// LSTM_55448027791477
// MI455X (gfx1250) — matrix operand order repaired and checked by reading, not compiled, not run on hardware
//
#include <hip/hip_runtime.h>
#include <hip/hip_bf16.h>
#include <math.h>

// ---------------------------------------------------------------------------
// LSTM policy net on MI455X (gfx1250, wave32, WMMA bf16 16x16x32)
//
//   Phase 0: repack Wx/Wh/Wa (f32 row-major) -> bf16 WMMA B-fragment layout
//   Phase 1: x_proj = inputs @ Wx + b : bf16 WMMA GEMM with the per-ktile B
//            slice staged into LDS (double-buffered async-to-LDS copies) so
//            WMMAs wait on ~30-cycle LDS reads instead of L2 round-trips.
//   Phase 2: serial scan, 16 WGs x 16 waves; Wh resident in VGPRs (256/wave),
//            Wa resident in LDS, h double-buffered in LDS A-fragment layout,
//            c in registers, gate-strided ntile ownership -> cell update is
//            wave-local; only 3 barriers per timestep.
// ---------------------------------------------------------------------------

typedef __attribute__((ext_vector_type(16))) __bf16 v16bf;
typedef __attribute__((ext_vector_type(2)))  __bf16 v2bf;
typedef __attribute__((ext_vector_type(8)))  float  v8f;
typedef __attribute__((ext_vector_type(4)))  int    v4i;

#define T_STEPS 512
#define B_TOT   256
#define D_IN    512
#define H_DIM   256
#define A_DIM   32
#define FOURH   1024
#define NT4H    64          // 4H / 16 n-tiles
#define KT_D    16          // D / 32 k-tiles
#define KT_H    8           // H / 32 k-tiles
#define MTILES  (T_STEPS * B_TOT / 16)   // 8192
#define HF_DW   (KT_H * 32 * 8)          // 2048 dwords per h-fragment buffer
#define WA_DWORDS (2 * KT_H * 32 * 8)    // 4096 dwords = 16 KB packed Wa

// ---- bf16 helpers (round-to-nearest-even) ---------------------------------
__device__ __forceinline__ unsigned bf16r(float f) {
  unsigned u = __float_as_uint(f);
  return (u + 0x7FFFu + ((u >> 16) & 1u)) >> 16;
}
__device__ __forceinline__ float bf16f(unsigned h) {
  return __uint_as_float(h << 16);
}
__device__ __forceinline__ unsigned pk2(float lo, float hi) {
#if __has_builtin(__builtin_amdgcn_cvt_pk_bf16_f32)
  v2bf p = __builtin_amdgcn_cvt_pk_bf16_f32(lo, hi);
  return __builtin_bit_cast(unsigned, p);
#else
  return bf16r(lo) | (bf16r(hi) << 16);
#endif
}
// A-fragment K offset for (vgpr j, lane-half): ISA 7.12.2 16-bit A 16x32
__device__ __forceinline__ int kbase(int j, int half) {
  return ((j < 4) ? (2 * j) : (16 + 2 * (j - 4))) + 8 * half;
}
__device__ __forceinline__ float sigf(float x) {
  return 1.0f / (1.0f + __expf(-x));
}

// ---- async global->LDS copy (CDNA5 GLOBAL_LOAD_ASYNC_TO_LDS_B128) ---------
__device__ __forceinline__ void cp_b128(const uint4* g, uint4* l) {
#if __has_builtin(__builtin_amdgcn_global_load_async_to_lds_b128)
  __builtin_amdgcn_global_load_async_to_lds_b128(
      (__attribute__((address_space(1))) v4i*)g,
      (__attribute__((address_space(3))) v4i*)l, 0, 0);
#else
  *l = *g;   // sync fallback: global_load_b128 + ds_store_b128
#endif
}
__device__ __forceinline__ void cp_wait() {
#if __has_builtin(__builtin_amdgcn_global_load_async_to_lds_b128)
#if __has_builtin(__builtin_amdgcn_s_wait_asynccnt)
  __builtin_amdgcn_s_wait_asynccnt(0);
#else
  asm volatile("s_wait_asynccnt 0x0" ::: "memory");
#endif
#endif
}

// ---------------------------------------------------------------------------
// Phase 0: pack W[K,N] f32 -> bf16 B-fragments [nt][kt][lane][8 dwords]
//   VGPR j, lanes 0-15: K = 2j,2j+1 (j<4), 8+2j,9+2j (j>=4) ; lanes 16-31: same +8 (per ktile)
// ---------------------------------------------------------------------------
__global__ void pack_bmat(const float* __restrict__ W, unsigned* __restrict__ out,
                          int K, int N) {
  int idx = blockIdx.x * blockDim.x + threadIdx.x;
  int total = (N / 16) * (K / 32) * 32 * 8;
  if (idx >= total) return;
  int j    = idx & 7;
  int lane = (idx >> 3) & 31;
  int kt   = (idx >> 8) % (K / 32);
  int nt   = (idx >> 8) / (K / 32);
  int n = nt * 16 + (lane & 15);
  int k = kt * 32 + kbase(j, lane >> 4);
  out[idx] = pk2(W[(size_t)k * N + n], W[(size_t)(k + 1) * N + n]);
}

// ---------------------------------------------------------------------------
// Phase 1: x_proj GEMM.  grid = (MTILES/8, 4), block = 256 (8 waves).
// wave w: M-tile = bx*8+w; the 16 n-tiles (ntbase..+15) are shared by all
// waves, so the per-ktile 16 KB B slice is staged into LDS (double-buffered).
// Output bf16 packed [mtile][nt][lane][8 bf16] = C-fragment order r=0..7.
// ---------------------------------------------------------------------------
__global__ void __launch_bounds__(256)
xproj_gemm(const float* __restrict__ X, const unsigned* __restrict__ WxP,
           const float* __restrict__ bias, uint4* __restrict__ XP) {
  __shared__ __align__(16) uint4 bbuf[2][1024];   // 2 x 16 KB B slices
  const int tid  = threadIdx.x;
  const int w    = tid >> 5, lane = tid & 31;
  const int half = lane >> 4, row = lane & 15;
  const int mtile  = blockIdx.x * 8 + w;
  const int ntbase = blockIdx.y * 16;
  const float* Xrow = X + (size_t)(mtile * 16 + row) * D_IN;
  const uint4* BPq = (const uint4*)WxP;           // frag (nt,kt) = 64 uint4

  v8f acc[16];
#pragma unroll
  for (int nt = 0; nt < 16; ++nt) {
    float bb = bias[(ntbase + nt) * 16 + row];
#pragma unroll
    for (int r = 0; r < 8; ++r) acc[nt][r] = bb;
  }

  // stage kt = 0
#pragma unroll
  for (int rnd = 0; rnd < 4; ++rnd) {
    int e = tid + rnd * 256;                       // 0..1023 16B units
    int ntl = e >> 6, u = e & 63;
    cp_b128(BPq + ((size_t)(ntbase + ntl) * KT_D + 0) * 64 + u, &bbuf[0][e]);
  }
  cp_wait();
  __syncthreads();

  int buf = 0;
  for (int kt = 0; kt < KT_D; ++kt) {
    if (kt + 1 < KT_D) {                           // prefetch next B slice
#pragma unroll
      for (int rnd = 0; rnd < 4; ++rnd) {
        int e = tid + rnd * 256;
        int ntl = e >> 6, u = e & 63;
        cp_b128(BPq + ((size_t)(ntbase + ntl) * KT_D + (kt + 1)) * 64 + u,
                &bbuf[buf ^ 1][e]);
      }
    }
    union { v16bf v; unsigned u[8]; } a;           // A fragment: f32 -> bf16
#pragma unroll
    for (int j = 0; j < 8; ++j) {
      const float2 f = *(const float2*)(Xrow + kt * 32 + kbase(j, half));
      a.u[j] = pk2(f.x, f.y);
    }
#pragma unroll
    for (int nt = 0; nt < 16; ++nt) {
      union { v16bf v; uint4 q[2]; } bw;
      bw.q[0] = bbuf[buf][nt * 64 + lane * 2 + 0];
      bw.q[1] = bbuf[buf][nt * 64 + lane * 2 + 1];
      acc[nt] = __builtin_amdgcn_wmma_f32_16x16x32_bf16(
          false, a.v, false, bw.v, (short)0, acc[nt], false, false);
    }
    cp_wait();          // next slice landed in LDS
    __syncthreads();    // everyone done reading current slice
    buf ^= 1;
  }
#pragma unroll
  for (int nt = 0; nt < 16; ++nt) {
    uint4 q;
    q.x = pk2(acc[nt][0], acc[nt][1]);
    q.y = pk2(acc[nt][2], acc[nt][3]);
    q.z = pk2(acc[nt][4], acc[nt][5]);
    q.w = pk2(acc[nt][6], acc[nt][7]);
    XP[((size_t)mtile * NT4H + (ntbase + nt)) * 32 + lane] = q;
  }
}

// ---------------------------------------------------------------------------
// Phase 2: the scan.  grid = 16 (batch tiles), block = 512 (16 waves).
// Wave w owns columns [16w,16w+16) of ALL FOUR gates (ntiles w,16+w,32+w,48+w)
// so i/f/g/o for its columns live in its own accumulators.  h fragments are
// double-buffered in LDS -> only 3 barriers per timestep.
// ---------------------------------------------------------------------------
__global__ void __launch_bounds__(512, 1)
lstm_scan(const uint4* __restrict__ XP, const unsigned* __restrict__ WhP,
          const unsigned* __restrict__ WaP, const int* __restrict__ new_ep,
          const float* __restrict__ carry_c, const float* __restrict__ carry_h,
          const float* __restrict__ ba, const float* __restrict__ Wv,
          const float* __restrict__ bvp, float* __restrict__ out) {
  __shared__ __align__(16) unsigned hfrag[2][HF_DW];  // h, A-frag layout (2x8 KB)
  __shared__ __align__(16) unsigned waLds[WA_DWORDS]; // packed Wa (16 KB)
  __shared__ int   resetf[16];
  __shared__ float vacc[16];                           // value-head accum

  const int tid  = threadIdx.x;
  const int w    = tid >> 5, lane = tid & 31;
  const int half = lane >> 4, row = lane & 15;
  const int mtile = blockIdx.x;
  const int b0    = mtile * 16;

  // --- Wh resident in VGPRs: 4 gate n-tiles x 8 k-tiles (256 VGPRs) -------
  v16bf whB[4][KT_H];
#pragma unroll
  for (int g4 = 0; g4 < 4; ++g4)
#pragma unroll
    for (int kt = 0; kt < KT_H; ++kt)
      whB[g4][kt] =
          ((const v16bf*)WhP)[((size_t)(g4 * 16 + w) * KT_H + kt) * 32 + lane];

  // --- c state in registers: element (row m = r+8*half, col jcol) ----------
  const int jcol = 16 * w + row;
  float c[8];
#pragma unroll
  for (int r = 0; r < 8; ++r)
    c[r] = carry_c[(size_t)(b0 + r + 8 * half) * H_DIM + jcol];

  // scatter coordinates of column jcol inside the A-fragment layout
  const int hkt  = jcol >> 5;
  const int k32  = jcol & 31;
  const int hl   = (k32 >> 3) & 1;
  const int k32p = k32 - 8 * hl;
  const int jj   = (k32p < 8) ? (k32p >> 1) : (4 + ((k32p - 16) >> 1));
  const int sub  = k32p & 1;

  const float mu_bias = (w < 2) ? ba[w * 16 + row] : 0.0f;
  const float bval    = bvp[0];
  const float wvc     = Wv[jcol];

  // --- init: h fragments from carry_h, Wa -> LDS, flags/value for t=0 ------
  for (int d = tid; d < HF_DW; d += 512) {
    int j = d & 7, ln = (d >> 3) & 31, kt = d >> 8;
    int k = kt * 32 + kbase(j, ln >> 4);
    const float* hr = carry_h + (size_t)(b0 + (ln & 15)) * H_DIM;
    hfrag[0][d] = pk2(hr[k], hr[k + 1]);
  }
  for (int d = tid; d < WA_DWORDS; d += 512) waLds[d] = WaP[d];
  if (tid < 16) {
    resetf[tid] = new_ep[b0 + tid];
    vacc[tid]   = bval;
  }
  __syncthreads();

  for (int t = 0; t < T_STEPS; ++t) {
    unsigned* rb = hfrag[t & 1];              // h(t) read buffer
    unsigned* wb = hfrag[(t + 1) & 1];        // new_h write buffer

    // ---- episode reset: zero h rows in rb, zero c (flags prefetched) ------
    for (int d = tid; d < HF_DW; d += 512)
      if (resetf[(d >> 3) & 15]) rb[d] = 0u;  // A-frag row = lane&15
#pragma unroll
    for (int r = 0; r < 8; ++r)
      if (resetf[r + 8 * half]) c[r] = 0.0f;
    __syncthreads();                           // (A)

    // ---- z = x_proj(t) + h @ Wh  (4 gate tiles per wave) ------------------
    const uint4* xpb = XP + (size_t)(t * 16 + mtile) * NT4H * 32;
    v8f z[4];
#pragma unroll
    for (int g4 = 0; g4 < 4; ++g4) {
      uint4 q = xpb[(size_t)(g4 * 16 + w) * 32 + lane];
      z[g4][0] = bf16f(q.x & 0xFFFFu); z[g4][1] = bf16f(q.x >> 16);
      z[g4][2] = bf16f(q.y & 0xFFFFu); z[g4][3] = bf16f(q.y >> 16);
      z[g4][4] = bf16f(q.z & 0xFFFFu); z[g4][5] = bf16f(q.z >> 16);
      z[g4][6] = bf16f(q.w & 0xFFFFu); z[g4][7] = bf16f(q.w >> 16);
    }
    if (t + 1 < T_STEPS)  // pull the adjacent x_proj m-tile (t*16+mtile+1) toward the WGP
      __builtin_prefetch((const void*)(xpb + (size_t)NT4H * 32), 0, 1);

#pragma unroll
    for (int kt = 0; kt < KT_H; ++kt) {
      union { v16bf v; uint4 q[2]; } a;
      const uint4* hp4 = (const uint4*)rb;
      a.q[0] = hp4[(kt * 32 + lane) * 2 + 0];
      a.q[1] = hp4[(kt * 32 + lane) * 2 + 1];
#pragma unroll
      for (int g4 = 0; g4 < 4; ++g4)
        z[g4] = __builtin_amdgcn_wmma_f32_16x16x32_bf16(
            false, a.v, false, whB[g4][kt], (short)0, z[g4], false, false);
    }
    // no barrier: cell writes go to the other buffer

    // ---- LSTM cell in registers; scatter new_h into wb --------------------
    unsigned short* hfh = (unsigned short*)wb;
    float pv[8];
#pragma unroll
    for (int r = 0; r < 8; ++r) {
      float iv = sigf(z[0][r]);
      float fv = sigf(z[1][r]);
      float gv = tanhf(z[2][r]);
      float ov = sigf(z[3][r]);
      float nc = fv * c[r] + iv * gv;
      c[r] = nc;
      float nh = ov * tanhf(nc);
      int m = r + 8 * half;
      hfh[((hkt * 32 + hl * 16 + m) * 8 + jj) * 2 + sub] = (unsigned short)bf16r(nh);
      pv[r] = nh * wvc;  // value-head partial
    }
    // value head: reduce over the 16 lanes of each half, then LDS float atomics
#pragma unroll
    for (int msk = 1; msk < 16; msk <<= 1)
#pragma unroll
      for (int r = 0; r < 8; ++r) pv[r] += __shfl_xor(pv[r], msk, 32);
    if (row == 0) {
#pragma unroll
      for (int r = 0; r < 8; ++r) atomicAdd(&vacc[r + 8 * half], pv[r]);
    }
    __syncthreads();                           // (C) new_h complete

    // ---- output heads; prefetch next step's flags; re-arm value accum -----
    float* outp = out + (size_t)(t * B_TOT + b0) * (A_DIM + 1);
    if (w < 2) {  // mu = new_h @ Wa + ba   (wave-uniform branch: EXEC all 1s)
      v8f macc;
#pragma unroll
      for (int r = 0; r < 8; ++r) macc[r] = mu_bias;
#pragma unroll
      for (int kt = 0; kt < KT_H; ++kt) {
        union { v16bf v; uint4 q[2]; } a;
        const uint4* hp4 = (const uint4*)wb;
        a.q[0] = hp4[(kt * 32 + lane) * 2 + 0];
        a.q[1] = hp4[(kt * 32 + lane) * 2 + 1];
        union { v16bf v; uint4 q[2]; } bwa;
        const uint4* wp4 = (const uint4*)waLds;
        bwa.q[0] = wp4[((size_t)(w * KT_H + kt) * 32 + lane) * 2 + 0];
        bwa.q[1] = wp4[((size_t)(w * KT_H + kt) * 32 + lane) * 2 + 1];
        macc = __builtin_amdgcn_wmma_f32_16x16x32_bf16(
            false, a.v, false, bwa.v, (short)0, macc, false, false);
      }
#pragma unroll
      for (int r = 0; r < 8; ++r)
        outp[(r + 8 * half) * (A_DIM + 1) + w * 16 + row] = macc[r];
    }
    if (tid < 16) {
      outp[tid * (A_DIM + 1) + A_DIM] = vacc[tid];
      vacc[tid] = bval;                        // re-arm for t+1
    } else if (tid < 32 && t + 1 < T_STEPS) {  // prefetch reset flags for t+1
      resetf[tid - 16] = new_ep[(t + 1) * B_TOT + b0 + (tid - 16)];
    }
    __syncthreads();                           // (D) end of step
  }
}

// ---------------------------------------------------------------------------
extern "C" void kernel_launch(void* const* d_in, const int* in_sizes, int n_in,
                              void* d_out, int out_size, void* d_ws, size_t ws_size,
                              hipStream_t stream) {
  (void)in_sizes; (void)n_in; (void)out_size; (void)ws_size;
  const float* inputs  = (const float*)d_in[0];   // [T,B,D]
  const int*   new_ep  = (const int*)  d_in[1];   // [T,B,1]
  const float* carry_c = (const float*)d_in[2];   // [B,H]
  const float* carry_h = (const float*)d_in[3];   // [B,H]
  const float* Wx      = (const float*)d_in[4];   // [D,4H]
  const float* Wh      = (const float*)d_in[5];   // [H,4H]
  const float* bias    = (const float*)d_in[6];   // [4H]
  const float* Wa      = (const float*)d_in[7];   // [H,A]
  const float* ba      = (const float*)d_in[8];   // [A]
  const float* Wv      = (const float*)d_in[9];   // [H,1]
  const float* bv      = (const float*)d_in[10];  // [1]
  float* out = (float*)d_out;                     // [T,B,A+1]

  // workspace carve-up (bf16 packed weights + bf16 x_proj fragments)
  const size_t WX_DW = (size_t)NT4H * KT_D * 32 * 8;  // 262144 dwords (1 MB)
  const size_t WH_DW = (size_t)NT4H * KT_H * 32 * 8;  // 131072 dwords (512 KB)
  const size_t WA_DW = (size_t)WA_DWORDS;             // 4096 dwords (16 KB)
  unsigned* wxP = (unsigned*)d_ws;
  unsigned* whP = wxP + WX_DW;
  unsigned* waP = whP + WH_DW;
  uint4*    xpP = (uint4*)(waP + WA_DW);              // 268 MB x_proj fragments

  pack_bmat<<<(int)((WX_DW + 255) / 256), 256, 0, stream>>>(Wx, wxP, D_IN, FOURH);
  pack_bmat<<<(int)((WH_DW + 255) / 256), 256, 0, stream>>>(Wh, whP, H_DIM, FOURH);
  pack_bmat<<<(int)((WA_DW + 255) / 256), 256, 0, stream>>>(Wa, waP, H_DIM, A_DIM);

  xproj_gemm<<<dim3(MTILES / 8, NT4H / 16), 256, 0, stream>>>(inputs, wxP, bias, xpP);

  lstm_scan<<<B_TOT / 16, 512, 0, stream>>>(xpP, whP, waP, new_ep, carry_c,
                                            carry_h, ba, Wv, bv, out);
}
